// DeepAR_41884521070850
// MI455X (gfx1250) — compile-verified, matrix operand order checked by reading, not run on hardware
//
#include <hip/hip_runtime.h>
#include <math.h>
#include <stdint.h>

typedef __attribute__((ext_vector_type(16))) _Float16 v16h;
typedef __attribute__((ext_vector_type(8)))  _Float16 v8h;
typedef __attribute__((ext_vector_type(8)))  float    v8f;

#define B_     512
#define SEQL   192
#define PREDL  96
#define WIN_   288
#define HID_   128
#define NGATE  512      // 4*HID
#define K0P    96       // input feature width 67 padded to multiple of 32

// ---- workspace layout (bytes) ----
#define XH_OFF    0
#define XH_BYTES  ((size_t)WIN_ * B_ * K0P * 2)          // 28,311,552
#define WIH0_OFF  (XH_OFF + XH_BYTES)
#define WIH0_BYTES ((size_t)NGATE * K0P * 2)             // 98,304
#define WHH0_OFF  (WIH0_OFF + WIH0_BYTES)
#define WMAT_BYTES ((size_t)NGATE * HID_ * 2)            // 131,072
#define WIH1_OFF  (WHH0_OFF + WMAT_BYTES)
#define WHH1_OFF  (WIH1_OFF + WMAT_BYTES)
#define B0_OFF    (WHH1_OFF + WMAT_BYTES)
#define B1_OFF    (B0_OFF + NGATE * 4)

__device__ __forceinline__ float sigf(float x) { return 1.0f / (1.0f + __expf(-x)); }

// ---------------------------------------------------------------------------
// Kernel 1: gather embeddings + assemble padded f16 input matrix Xh[t][b][K0P]
// ---------------------------------------------------------------------------
__global__ void assemble_inputs(const float* __restrict__ given_enc,   // (B,W,7)
                                const float* __restrict__ x_enc,       // (B,192,1)
                                const float* __restrict__ x_mark,      // (B,W,4)
                                const float* __restrict__ meta_x,      // (B,4)
                                const float* __restrict__ batch_y,     // (B,96,1)
                                const float* __restrict__ te0,         // (31,8)
                                const float* __restrict__ te1,         // (12,8)
                                const float* __restrict__ te2,         // (7,8)
                                const float* __restrict__ me0,         // (1000,16)
                                const float* __restrict__ me1,         // (100,16)
                                _Float16* __restrict__ Xh) {
    int row = blockIdx.x * blockDim.x + threadIdx.x;
    if (row >= B_ * WIN_) return;
    int t = row / B_;
    int b = row - t * B_;
    _Float16* dst = Xh + (size_t)row * K0P;

    // autoregressive previous target (teacher forced)
    float prev;
    if (t == 0)            prev = x_enc[(size_t)b * SEQL];
    else if (t <= SEQL)    prev = x_enc[(size_t)b * SEQL + (t - 1)];
    else                   prev = batch_y[(size_t)b * PREDL + (t - SEQL - 1)];
    dst[0] = (_Float16)prev;

    const float* ge = given_enc + ((size_t)b * WIN_ + t) * 7;
    #pragma unroll
    for (int i = 0; i < 4; ++i) dst[1 + i] = (_Float16)ge[i];
    int id0 = (int)ge[4], id1 = (int)ge[5], id2 = (int)ge[6];
    #pragma unroll
    for (int i = 0; i < 8; ++i) dst[5  + i] = (_Float16)te0[id0 * 8 + i];
    #pragma unroll
    for (int i = 0; i < 8; ++i) dst[13 + i] = (_Float16)te1[id1 * 8 + i];
    #pragma unroll
    for (int i = 0; i < 8; ++i) dst[21 + i] = (_Float16)te2[id2 * 8 + i];

    const float* xm = x_mark + ((size_t)b * WIN_ + t) * 4;
    #pragma unroll
    for (int i = 0; i < 4; ++i) dst[29 + i] = (_Float16)xm[i];

    const float* mx = meta_x + (size_t)b * 4;
    dst[33] = (_Float16)mx[0];
    dst[34] = (_Float16)mx[1];
    int mid0 = (int)mx[2], mid1 = (int)mx[3];
    #pragma unroll
    for (int i = 0; i < 16; ++i) dst[35 + i] = (_Float16)me0[mid0 * 16 + i];
    #pragma unroll
    for (int i = 0; i < 16; ++i) dst[51 + i] = (_Float16)me1[mid1 * 16 + i];
    for (int i = 67; i < K0P; ++i) dst[i] = (_Float16)0.0f;
}

// ---------------------------------------------------------------------------
// Kernel 2: cast weights to f16 (W_ih0 zero-padded 67->96), pre-sum bias pairs
// ---------------------------------------------------------------------------
__global__ void prep_weights(const float* __restrict__ Wih0, const float* __restrict__ Whh0,
                             const float* __restrict__ Wih1, const float* __restrict__ Whh1,
                             const float* __restrict__ bih0, const float* __restrict__ bhh0,
                             const float* __restrict__ bih1, const float* __restrict__ bhh1,
                             _Float16* __restrict__ Wih0h, _Float16* __restrict__ Whh0h,
                             _Float16* __restrict__ Wih1h, _Float16* __restrict__ Whh1h,
                             float* __restrict__ b0, float* __restrict__ b1) {
    int i = blockIdx.x * blockDim.x + threadIdx.x;
    if (i < NGATE * K0P) {
        int n = i / K0P, k = i - n * K0P;
        Wih0h[i] = (_Float16)((k < 67) ? Wih0[(size_t)n * 67 + k] : 0.0f);
    }
    if (i < NGATE * HID_) {
        Whh0h[i] = (_Float16)Whh0[i];
        Wih1h[i] = (_Float16)Wih1[i];
        Whh1h[i] = (_Float16)Whh1[i];
    }
    if (i < NGATE) {
        b0[i] = bih0[i] + bhh0[i];
        b1[i] = bih1[i] + bhh1[i];
    }
}

// ---------------------------------------------------------------------------
// A/B fragment loader for v_wmma_f32_16x16x32_f16, wave32 layout (ISA 7.12.2):
// lane L holds row (L&15); lanes 0-15 get K = kk+[0..7] and kk+[16..23],
// lanes 16-31 get K = kk+[8..15] and kk+[24..31]. Two 16-byte chunks.
// Works for B fragments too: weight copies are stored (n,k) row-major = B^T.
// ---------------------------------------------------------------------------
__device__ __forceinline__ v16h load_frag16(const _Float16* base, int stride, int lane, int kk) {
    int m  = lane & 15;
    int ks = (lane >> 4) << 3;
    const _Float16* p = base + (size_t)m * stride + kk + ks;
    v8h lo = *(const v8h*)(p);
    v8h hi = *(const v8h*)(p + 16);
    v16h r;
    #pragma unroll
    for (int i = 0; i < 8; ++i) { r[i] = lo[i]; r[i + 8] = hi[i]; }
    return r;
}

// ---------------------------------------------------------------------------
// Kernel 3: persistent 2-layer LSTM. One block = 16 batch rows x all 288 steps.
// 256 threads = 8 waves (2 waves/SIMD -> ~512 VGPR/wave budget).
// Wave w owns gate columns [w*64, w*64+64) = 4 WMMA N-tiles.
// Recurrent weights W_hh0/W_hh1 (32 frags = 256 VGPRs) live in registers for
// the whole loop; W_ih0/W_ih1 are streamed from L2 each step. The streamed
// bases are offset by a loop-variant SGPR laundered through empty inline asm:
// this defeats LICM hoist-and-spill while keeping global-pointer provenance
// (so the loads stay global_load_b128, not flat_load, and use only LOADcnt).
// ---------------------------------------------------------------------------
__launch_bounds__(256, 1)
__global__ void lstm_persist(const _Float16* __restrict__ Xh,
                             const _Float16* __restrict__ Wih0,
                             const _Float16* __restrict__ Whh0,
                             const _Float16* __restrict__ Wih1,
                             const _Float16* __restrict__ Whh1,
                             const float* __restrict__ b0,
                             const float* __restrict__ b1,
                             const float* __restrict__ Wmean,
                             const float* __restrict__ bmean,
                             float* __restrict__ out) {
    __shared__ __align__(16) float    zbuf[16][NGATE];   // 32 KB gate pre-activations
    __shared__ __align__(16) _Float16 hb1[16][HID_];     // 4 KB  h1 (f16, A-matrix feed)
    __shared__ __align__(16) _Float16 hb2[16][HID_];     // 4 KB  h2 (f16, A-matrix feed)
    __shared__ __align__(16) float    h2f[16][HID_];     // 8 KB  h2 (f32, output head)
    __shared__ __align__(16) float    wmn[HID_];         // 512 B output head weights

    const int tid  = threadIdx.x;
    const int lane = tid & 31;
    const int wv   = tid >> 5;          // wave 0..7
    const int gb0  = blockIdx.x * 16;   // global batch base of this tile

    // zero hidden state; stage W_mean
    for (int i = tid; i < 16 * HID_; i += 256) {
        (&hb1[0][0])[i] = (_Float16)0.0f;
        (&hb2[0][0])[i] = (_Float16)0.0f;
    }
    if (tid < HID_) wmn[tid] = Wmean[tid];

    // per-thread cell state: thread owns row cm, hidden units cj0..cj0+7
    const int cm  = tid >> 4;           // 0..15
    const int cj0 = (tid & 15) << 3;    // 0..120
    float c1r[8], c2r[8];
    #pragma unroll
    for (int q = 0; q < 8; ++q) { c1r[q] = 0.0f; c2r[q] = 0.0f; }

    const int N0 = wv * 64;       // this wave's 64-col gate strip (4 tiles)
    const int zm = lane >> 4;     // C/D layout: VGPR r -> row r + 8*zm
    const int zn = lane & 15;     // C/D layout: lane -> column

    // ---- register-resident recurrent weights: 32 frags = 256 VGPRs ----
    v16h wHH0[16], wHH1[16];
    #pragma unroll
    for (int nt = 0; nt < 4; ++nt)
        #pragma unroll
        for (int s = 0; s < 4; ++s) {
            wHH0[nt * 4 + s] = load_frag16(Whh0 + (size_t)(N0 + nt * 16) * HID_, HID_, lane, s * 32);
            wHH1[nt * 4 + s] = load_frag16(Whh1 + (size_t)(N0 + nt * 16) * HID_, HID_, lane, s * 32);
        }

    // per-lane gate bias for this wave's 4 tiles
    float b0v[4], b1v[4];
    #pragma unroll
    for (int nt = 0; nt < 4; ++nt) {
        b0v[nt] = b0[N0 + nt * 16 + zn];
        b1v[nt] = b1[N0 + nt * 16 + zn];
    }
    const float bm = bmean[0];

    __syncthreads();

    for (int t = 0; t < WIN_; ++t) {
        // opaque loop-variant offset (always 0): blocks LICM on streamed
        // weight loads without destroying global address-space provenance
        uint32_t zoff = 0;
        asm volatile("" : "+s"(zoff));
        const _Float16* w0 = Wih0 + zoff;
        const _Float16* w1 = Wih1 + zoff;

        // ======================= layer 0 =======================
        v8f acc[4];
        #pragma unroll
        for (int nt = 0; nt < 4; ++nt)
            #pragma unroll
            for (int r = 0; r < 8; ++r) acc[nt][r] = 0.0f;

        const _Float16* xrow = Xh + ((size_t)t * B_ + gb0) * K0P;
        if (t + 1 < WIN_)   // overlap next step's input traffic
            __builtin_prefetch(Xh + ((size_t)(t + 1) * B_ + gb0) * K0P + lane * 8, 0, 0);

        // input contribution: (16 x 96) @ (96 x 64), B streamed from L2
        #pragma unroll
        for (int s = 0; s < 3; ++s) {
            v16h a = load_frag16(xrow, K0P, lane, s * 32);
            #pragma unroll
            for (int nt = 0; nt < 4; ++nt) {
                v16h bf = load_frag16(w0 + (size_t)(N0 + nt * 16) * K0P, K0P, lane, s * 32);
                acc[nt] = __builtin_amdgcn_wmma_f32_16x16x32_f16(
                    false, a, false, bf, (short)0, acc[nt], false, false);
            }
        }
        // recurrent contribution: (16 x 128) @ (128 x 64), B in registers
        #pragma unroll
        for (int s = 0; s < 4; ++s) {
            v16h a = load_frag16(&hb1[0][0], HID_, lane, s * 32);
            #pragma unroll
            for (int nt = 0; nt < 4; ++nt)
                acc[nt] = __builtin_amdgcn_wmma_f32_16x16x32_f16(
                    false, a, false, wHH0[nt * 4 + s], (short)0, acc[nt], false, false);
        }
        // stage z (+bias) to LDS
        #pragma unroll
        for (int nt = 0; nt < 4; ++nt) {
            int n = N0 + nt * 16 + zn;
            #pragma unroll
            for (int r = 0; r < 8; ++r) zbuf[r + 8 * zm][n] = acc[nt][r] + b0v[nt];
        }
        __syncthreads();

        // LSTM cell, layer 0 (gate order i,f,g,o): 8 cells per thread
        #pragma unroll
        for (int q = 0; q < 8; ++q) {
            int j = cj0 + q;
            float zi = zbuf[cm][j];
            float zf = zbuf[cm][j + 128];
            float zg = zbuf[cm][j + 256];
            float zo = zbuf[cm][j + 384];
            float c  = sigf(zf) * c1r[q] + sigf(zi) * tanhf(zg);
            c1r[q]   = c;
            hb1[cm][j] = (_Float16)(sigf(zo) * tanhf(c));
        }
        __syncthreads();

        // ======================= layer 1 =======================
        #pragma unroll
        for (int nt = 0; nt < 4; ++nt)
            #pragma unroll
            for (int r = 0; r < 8; ++r) acc[nt][r] = 0.0f;

        // h1 contribution, B streamed from L2
        #pragma unroll
        for (int s = 0; s < 4; ++s) {
            v16h a = load_frag16(&hb1[0][0], HID_, lane, s * 32);
            #pragma unroll
            for (int nt = 0; nt < 4; ++nt) {
                v16h bf = load_frag16(w1 + (size_t)(N0 + nt * 16) * HID_, HID_, lane, s * 32);
                acc[nt] = __builtin_amdgcn_wmma_f32_16x16x32_f16(
                    false, a, false, bf, (short)0, acc[nt], false, false);
            }
        }
        // h2 recurrent contribution, B in registers
        #pragma unroll
        for (int s = 0; s < 4; ++s) {
            v16h a = load_frag16(&hb2[0][0], HID_, lane, s * 32);
            #pragma unroll
            for (int nt = 0; nt < 4; ++nt)
                acc[nt] = __builtin_amdgcn_wmma_f32_16x16x32_f16(
                    false, a, false, wHH1[nt * 4 + s], (short)0, acc[nt], false, false);
        }
        #pragma unroll
        for (int nt = 0; nt < 4; ++nt) {
            int n = N0 + nt * 16 + zn;
            #pragma unroll
            for (int r = 0; r < 8; ++r) zbuf[r + 8 * zm][n] = acc[nt][r] + b1v[nt];
        }
        __syncthreads();

        // LSTM cell, layer 1
        #pragma unroll
        for (int q = 0; q < 8; ++q) {
            int j = cj0 + q;
            float zi = zbuf[cm][j];
            float zf = zbuf[cm][j + 128];
            float zg = zbuf[cm][j + 256];
            float zo = zbuf[cm][j + 384];
            float c  = sigf(zf) * c2r[q] + sigf(zi) * tanhf(zg);
            c2r[q]   = c;
            float h  = sigf(zo) * tanhf(c);
            hb2[cm][j] = (_Float16)h;
            h2f[cm][j] = h;
        }
        __syncthreads();

        // output head on the last PRED_LEN steps: out[b, t-192] = h2 . W_mean + b
        if (t >= SEQL && tid < 16) {
            float s = bm;
            for (int j = 0; j < HID_; ++j) s += h2f[tid][j] * wmn[j];
            out[(size_t)(gb0 + tid) * PREDL + (t - SEQL)] = s;
        }
    }
}

extern "C" void kernel_launch(void* const* d_in, const int* in_sizes, int n_in,
                              void* d_out, int out_size, void* d_ws, size_t ws_size,
                              hipStream_t stream) {
    const float* given_enc = (const float*)d_in[0];
    const float* x_enc     = (const float*)d_in[1];
    const float* x_mark    = (const float*)d_in[2];
    const float* meta_x    = (const float*)d_in[3];
    const float* batch_y   = (const float*)d_in[4];
    const float* te0       = (const float*)d_in[5];
    const float* te1       = (const float*)d_in[6];
    const float* te2       = (const float*)d_in[7];
    const float* me0       = (const float*)d_in[8];
    const float* me1       = (const float*)d_in[9];
    const float* W_ih0     = (const float*)d_in[10];
    const float* W_hh0     = (const float*)d_in[11];
    const float* b_ih0     = (const float*)d_in[12];
    const float* b_hh0     = (const float*)d_in[13];
    const float* W_ih1     = (const float*)d_in[14];
    const float* W_hh1     = (const float*)d_in[15];
    const float* b_ih1     = (const float*)d_in[16];
    const float* b_hh1     = (const float*)d_in[17];
    const float* W_mean    = (const float*)d_in[18];
    const float* b_mean    = (const float*)d_in[19];

    char* ws = (char*)d_ws;
    _Float16* Xh    = (_Float16*)(ws + XH_OFF);
    _Float16* Wih0h = (_Float16*)(ws + WIH0_OFF);
    _Float16* Whh0h = (_Float16*)(ws + WHH0_OFF);
    _Float16* Wih1h = (_Float16*)(ws + WIH1_OFF);
    _Float16* Whh1h = (_Float16*)(ws + WHH1_OFF);
    float*    b0    = (float*)(ws + B0_OFF);
    float*    b1    = (float*)(ws + B1_OFF);

    int rows = B_ * WIN_;
    assemble_inputs<<<(rows + 127) / 128, 128, 0, stream>>>(
        given_enc, x_enc, x_mark, meta_x, batch_y, te0, te1, te2, me0, me1, Xh);

    prep_weights<<<(NGATE * HID_ + 255) / 256, 256, 0, stream>>>(
        W_ih0, W_hh0, W_ih1, W_hh1, b_ih0, b_hh0, b_ih1, b_hh1,
        Wih0h, Whh0h, Wih1h, Whh1h, b0, b1);

    lstm_persist<<<B_ / 16, 256, 0, stream>>>(
        Xh, Wih0h, Whh0h, Wih1h, Whh1h, b0, b1, W_mean, b_mean, (float*)d_out);
}
